// GraphTransformer_15676630630942
// MI455X (gfx1250) — matrix operand order repaired and checked by reading, not compiled, not run on hardware
//
#include <hip/hip_runtime.h>
#include <hip/hip_bf16.h>

// ---------------- problem constants ----------------
constexpr int Tt = 256, Ss = 256, Bb = 2, Dd = 512, Hh = 8, HDh = 64, FFf = 2048, Ll = 2;
constexpr float SCALING = 0.125f;   // HD^-0.5 = 64^-0.5

typedef __attribute__((ext_vector_type(16))) __bf16 v16bf;
typedef __attribute__((ext_vector_type(8)))  __bf16 v8bf;
typedef __attribute__((ext_vector_type(8)))  float  v8f;
typedef __attribute__((ext_vector_type(4)))  float  v4f;

static __device__ __forceinline__ __bf16 f2bf(float f) {
  union { float f; unsigned u; } v; v.f = f;
  unsigned r = v.u + 0x7FFFu + ((v.u >> 16) & 1u);   // round-to-nearest-even
  unsigned short h = (unsigned short)(r >> 16);
  return __builtin_bit_cast(__bf16, h);
}

static __device__ __forceinline__ v8f wmma_bf16(v16bf a, v16bf b, v8f c) {
  // (neg_a, A, neg_b, B, c_mod, C, reuse_a, reuse_b)
  return __builtin_amdgcn_wmma_f32_16x16x32_bf16(false, a, false, b, (short)0, c, false, false);
}

// ---------------- generic bf16 NT GEMM:  C[m,n] = sum_k A[m,k]*Bw[n,k] (+bias, relu) ---------
// block = 256 threads = 8 waves, each wave computes a 16(M) x 64(N) tile.
// wave arrangement: wm waves along M, wn along N (wm*wn == 8).
// mode 0: C index = m*ldc + n.  mode 1 (attention scatter, z = b*H+h): C index = (m*B+b)*D + h*64 + n.
__global__ void __launch_bounds__(256)
gemm_bf16_nt(const __bf16* __restrict__ A, int lda, long aStrideZ,
             const __bf16* __restrict__ Bw, int ldb, long bStrideZ,
             const float* __restrict__ bias,
             float* __restrict__ Cf, __bf16* __restrict__ Cb,
             int ldc, int K, int wm, int wn, int relu, int mode) {
  int wave = threadIdx.x >> 5;
  int lane = threadIdx.x & 31;
  int ll = lane & 15, lh = lane >> 4;
  int z = blockIdx.z;
  A  += (size_t)z * aStrideZ;
  Bw += (size_t)z * bStrideZ;
  int waveM = wave / wn, waveN = wave % wn;
  int m0 = (blockIdx.y * wm + waveM) * 16;
  int n0 = (blockIdx.x * wn + waveN) * 64;

  v8f acc[4];
#pragma unroll
  for (int ct = 0; ct < 4; ++ct) acc[ct] = (v8f){0,0,0,0,0,0,0,0};

  const __bf16* arow = A + (size_t)(m0 + ll) * lda;
  for (int k0 = 0; k0 < K; k0 += 32) {
    int kb = k0 + lh * 16;
    v8bf a0 = *(const v8bf*)(arow + kb);
    v8bf a1 = *(const v8bf*)(arow + kb + 8);
    v16bf a = __builtin_shufflevector(a0, a1, 0,1,2,3,4,5,6,7,8,9,10,11,12,13,14,15);
#pragma unroll
    for (int ct = 0; ct < 4; ++ct) {
      const __bf16* brow = Bw + (size_t)(n0 + ct * 16 + ll) * ldb + k0 + lh * 16;
      v16bf b = *(const v16bf*)brow;
      acc[ct] = wmma_bf16(a, b, acc[ct]);
    }
  }

  int b_ = z / Hh, h_ = z % Hh;
#pragma unroll
  for (int ct = 0; ct < 4; ++ct) {
    int n = n0 + ct * 16 + ll;
    float bs = bias ? bias[n] : 0.0f;
#pragma unroll
    for (int v = 0; v < 8; ++v) {
      int m = m0 + v + lh * 8;
      float val = acc[ct][v] + bs;
      if (relu) val = fmaxf(val, 0.0f);
      size_t idx;
      if (mode == 0) idx = (size_t)m * ldc + n;
      else           idx = ((size_t)m * Bb + b_) * Dd + h_ * HDh + n;
      if (Cf) Cf[idx] = val;
      if (Cb) Cb[idx] = f2bf(val);
    }
  }
}

// ---------------- fused relation projection + attention scores -------------------------------
// scores[b,h,t,s] = scaling * sum_hd (q[t,bh,hd] + ra[s,t,bh,hd]) * (k[s,bh,hd] + rb[s,t,bh,hd])
// where ra/rb = relation[s,t,b,:] @ rel_w[.]^T  (note the swapaxes(0,1) in the reference!)
// grid = (T, S/16, B), block = 256 (8 waves, wave == head). K loop is WMMA over D.
__global__ void __launch_bounds__(256)
rel_scores(const float* __restrict__ relation,   // (T,S,B,D) f32
           const __bf16* __restrict__ rwb,       // (2D, D) bf16 for this layer
           const float* __restrict__ qkv,        // (T*B, 3D) f32
           float* __restrict__ scores) {         // (B,H,T,S) f32
  int t = blockIdx.x;
  int s0 = blockIdx.y * 16;
  int b = blockIdx.z;
  int h = threadIdx.x >> 5;
  int lane = threadIdx.x & 31;
  int ll = lane & 15, lh = lane >> 4;

  v8f ra[4], rb[4];
#pragma unroll
  for (int ct = 0; ct < 4; ++ct) { ra[ct] = (v8f){0,0,0,0,0,0,0,0}; rb[ct] = ra[ct]; }

  // A row M=ll maps to source index s = s0 + M; relation indexed [s, t] (swapaxes in ref)
  const float* arow = relation + (((size_t)(s0 + ll) * Tt + t) * Bb + b) * Dd;

  for (int k0 = 0; k0 < Dd; k0 += 32) {
    __builtin_prefetch(arow + k0 + 64, 0, 0);
    int kb = k0 + lh * 16;
    float fl[16];
    *(v4f*)(fl + 0)  = *(const v4f*)(arow + kb);
    *(v4f*)(fl + 4)  = *(const v4f*)(arow + kb + 4);
    *(v4f*)(fl + 8)  = *(const v4f*)(arow + kb + 8);
    *(v4f*)(fl + 12) = *(const v4f*)(arow + kb + 12);
    v16bf a;
#pragma unroll
    for (int i = 0; i < 16; ++i) a[i] = f2bf(fl[i]);
#pragma unroll
    for (int ct = 0; ct < 4; ++ct) {
      const __bf16* bra = rwb + (size_t)(h * HDh + ct * 16 + ll) * Dd + k0 + lh * 16;
      const __bf16* brb = bra + (size_t)Dd * Dd;   // rb block starts at row D
      ra[ct] = wmma_bf16(a, *(const v16bf*)bra, ra[ct]);
      rb[ct] = wmma_bf16(a, *(const v16bf*)brb, rb[ct]);
    }
  }

  // epilogue: per-element (q+ra)*(k+rb), then row-sum over the head's 64 columns
  const float* qrow = qkv + ((size_t)t * Bb + b) * (3 * Dd);
  float qv[4];
#pragma unroll
  for (int ct = 0; ct < 4; ++ct) qv[ct] = qrow[h * HDh + ct * 16 + ll];

  float part[8];
#pragma unroll
  for (int v = 0; v < 8; ++v) {
    int s = s0 + v + lh * 8;
    const float* krow = qkv + ((size_t)s * Bb + b) * (3 * Dd) + Dd;
    float sum = 0.0f;
#pragma unroll
    for (int ct = 0; ct < 4; ++ct) {
      float kv = krow[h * HDh + ct * 16 + ll];
      sum += (qv[ct] + ra[ct][v]) * (kv + rb[ct][v]);
    }
    part[v] = sum;
  }
  // reduce over the 16 lanes of each half-wave (columns of the C tile)
#pragma unroll
  for (int m = 1; m < 16; m <<= 1)
#pragma unroll
    for (int v = 0; v < 8; ++v) part[v] += __shfl_xor(part[v], m, 32);

  if (ll == 0) {
    size_t base = (((size_t)(b * Hh + h) * Tt) + t) * Ss;
#pragma unroll
    for (int v = 0; v < 8; ++v)
      scores[base + s0 + v + lh * 8] = part[v] * SCALING;
  }
}

// ---------------- softmax over S, writes bf16 probabilities ----------------------------------
__global__ void __launch_bounds__(256)
softmax_rows(const float* __restrict__ scores, __bf16* __restrict__ P) {
  int row = blockIdx.x;           // b*H*T rows
  int tid = threadIdx.x;          // S == 256
  __shared__ float red[256];
  float v = scores[(size_t)row * Ss + tid];
  red[tid] = v; __syncthreads();
  for (int off = 128; off > 0; off >>= 1) { if (tid < off) red[tid] = fmaxf(red[tid], red[tid + off]); __syncthreads(); }
  float mx = red[0]; __syncthreads();
  float e = __expf(v - mx);
  red[tid] = e; __syncthreads();
  for (int off = 128; off > 0; off >>= 1) { if (tid < off) red[tid] += red[tid + off]; __syncthreads(); }
  float s = red[0];
  P[(size_t)row * Ss + tid] = f2bf(e / s);
}

// ---------------- V transpose: Vt[bh, hd, s] = v[s, bh, hd] (bf16) ---------------------------
__global__ void build_vt(const float* __restrict__ qkv, __bf16* __restrict__ Vt, int n) {
  int i = blockIdx.x * blockDim.x + threadIdx.x;
  if (i >= n) return;
  int s = i % Ss; int rest = i / Ss;
  int hd = rest % HDh; int bh = rest / HDh;
  int b = bh / Hh, h = bh % Hh;
  Vt[i] = f2bf(qkv[((size_t)s * Bb + b) * (3 * Dd) + 2 * Dd + h * HDh + hd]);
}

// ---------------- f32 -> bf16 cast -----------------------------------------------------------
__global__ void cast_f32_bf16(const float* __restrict__ src, __bf16* __restrict__ dst, int n) {
  int i = blockIdx.x * blockDim.x + threadIdx.x;
  if (i < n) dst[i] = f2bf(src[i]);
}

// ---------------- residual add + LayerNorm (writes f32 and bf16) -----------------------------
__global__ void __launch_bounds__(256)
add_layernorm(const float* __restrict__ resid, const float* __restrict__ y,
              const float* __restrict__ gamma, const float* __restrict__ beta,
              float* __restrict__ outf, __bf16* __restrict__ outb) {
  int row = blockIdx.x;            // T*B rows
  int tid = threadIdx.x;           // 256 threads, 2 elems each (D=512)
  __shared__ float red[256];
  size_t base = (size_t)row * Dd;
  float v0 = resid[base + tid]       + y[base + tid];
  float v1 = resid[base + tid + 256] + y[base + tid + 256];
  red[tid] = v0 + v1; __syncthreads();
  for (int off = 128; off > 0; off >>= 1) { if (tid < off) red[tid] += red[tid + off]; __syncthreads(); }
  float mean = red[0] * (1.0f / Dd); __syncthreads();
  float d0 = v0 - mean, d1 = v1 - mean;
  red[tid] = d0 * d0 + d1 * d1; __syncthreads();
  for (int off = 128; off > 0; off >>= 1) { if (tid < off) red[tid] += red[tid + off]; __syncthreads(); }
  float var = red[0] * (1.0f / Dd);
  float rstd = rsqrtf(var + 1e-5f);
  float o0 = gamma[tid]       * d0 * rstd + beta[tid];
  float o1 = gamma[tid + 256] * d1 * rstd + beta[tid + 256];
  outf[base + tid] = o0;
  outf[base + tid + 256] = o1;
  if (outb) { outb[base + tid] = f2bf(o0); outb[base + tid + 256] = f2bf(o1); }
}

// =============================================================================================
extern "C" void kernel_launch(void* const* d_in, const int* in_sizes, int n_in,
                              void* d_out, int out_size, void* d_ws, size_t ws_size,
                              hipStream_t stream) {
  const float* x_in  = (const float*)d_in[0];
  const float* rel   = (const float*)d_in[1];
  const float* in_w  = (const float*)d_in[2];
  const float* in_b  = (const float*)d_in[3];
  const float* rel_w = (const float*)d_in[4];
  const float* out_w = (const float*)d_in[5];
  const float* out_b = (const float*)d_in[6];
  const float* fc1_w = (const float*)d_in[7];
  const float* fc1_b = (const float*)d_in[8];
  const float* fc2_w = (const float*)d_in[9];
  const float* fc2_b = (const float*)d_in[10];
  const float* ln1_g = (const float*)d_in[11];
  const float* ln1_b = (const float*)d_in[12];
  const float* ln2_g = (const float*)d_in[13];
  const float* ln2_b = (const float*)d_in[14];

  // ---- carve workspace (256B aligned) ----
  char* p = (char*)d_ws;
  auto alloc = [&](size_t bytes) -> void* {
    void* r = (void*)p;
    p += (bytes + 255) & ~(size_t)255;
    return r;
  };
  const int NTB = Tt * Bb;                   // 512 rows
  __bf16* in_wb  = (__bf16*)alloc((size_t)Ll * 3 * Dd * Dd * 2);
  __bf16* rel_wb = (__bf16*)alloc((size_t)Ll * 2 * Dd * Dd * 2);
  __bf16* out_wb = (__bf16*)alloc((size_t)Ll * Dd * Dd * 2);
  __bf16* fc1_wb = (__bf16*)alloc((size_t)Ll * FFf * Dd * 2);
  __bf16* fc2_wb = (__bf16*)alloc((size_t)Ll * Dd * FFf * 2);
  __bf16* Xb     = (__bf16*)alloc((size_t)NTB * Dd * 2);
  float*  qkv    = (float*) alloc((size_t)NTB * 3 * Dd * 4);
  float*  scores = (float*) alloc((size_t)Bb * Hh * Tt * Ss * 4);
  __bf16* P      = (__bf16*)alloc((size_t)Bb * Hh * Tt * Ss * 2);
  __bf16* Vt     = (__bf16*)alloc((size_t)Bb * Hh * HDh * Ss * 2);
  float*  attnf  = (float*) alloc((size_t)NTB * Dd * 4);
  __bf16* attnb  = (__bf16*)alloc((size_t)NTB * Dd * 2);
  float*  proj   = (float*) alloc((size_t)NTB * Dd * 4);
  __bf16* hb     = (__bf16*)alloc((size_t)NTB * FFf * 2);
  float*  ffn    = (float*) alloc((size_t)NTB * Dd * 4);
  float*  curx   = (float*) alloc((size_t)NTB * Dd * 4);
  (void)attnf; (void)ws_size; (void)n_in; (void)in_sizes; (void)out_size;

  auto cast = [&](const float* s, __bf16* d, int n) {
    cast_f32_bf16<<<(n + 255) / 256, 256, 0, stream>>>(s, d, n);
  };
  // ---- pre-cast weights to bf16 (deterministic, every call) ----
  cast(in_w,  in_wb,  Ll * 3 * Dd * Dd);
  cast(rel_w, rel_wb, Ll * 2 * Dd * Dd);
  cast(out_w, out_wb, Ll * Dd * Dd);
  cast(fc1_w, fc1_wb, Ll * FFf * Dd);
  cast(fc2_w, fc2_wb, Ll * Dd * FFf);
  cast(x_in, Xb, NTB * Dd);

  for (int l = 0; l < Ll; ++l) {
    const float* xres = (l == 0) ? x_in : curx;
    const __bf16* inw_l  = in_wb  + (size_t)l * 3 * Dd * Dd;
    const __bf16* relw_l = rel_wb + (size_t)l * 2 * Dd * Dd;
    const __bf16* outw_l = out_wb + (size_t)l * Dd * Dd;
    const __bf16* fc1w_l = fc1_wb + (size_t)l * FFf * Dd;
    const __bf16* fc2w_l = fc2_wb + (size_t)l * Dd * FFf;

    // 1) qkv = X @ in_w^T + in_b  : M=512, N=1536, K=512
    gemm_bf16_nt<<<dim3(3 * Dd / 256, NTB / 32, 1), 256, 0, stream>>>(
        Xb, Dd, 0, inw_l, Dd, 0, in_b + (size_t)l * 3 * Dd,
        qkv, nullptr, 3 * Dd, Dd, /*wm*/2, /*wn*/4, 0, 0);

    // 2) fused relation projection + scores
    rel_scores<<<dim3(Tt, Ss / 16, Bb), 256, 0, stream>>>(rel, relw_l, qkv, scores);

    // 3) softmax over S -> bf16 probabilities
    softmax_rows<<<Bb * Hh * Tt, 256, 0, stream>>>(scores, P);

    // 4) Vt[bh,hd,s] = v[s,bh,hd]
    {
      int n = Bb * Hh * HDh * Ss;
      build_vt<<<(n + 255) / 256, 256, 0, stream>>>(qkv, Vt, n);
    }

    // 5) attn = P @ Vt^T per (b,h), scattered to (t,b,d): M=T, N=HD, K=S, Z=BH
    gemm_bf16_nt<<<dim3(1, Tt / 128, Bb * Hh), 256, 0, stream>>>(
        P, Ss, (long)Tt * Ss, Vt, Ss, (long)HDh * Ss, nullptr,
        attnf, attnb, 0, Ss, /*wm*/8, /*wn*/1, 0, /*mode*/1);

    // 6) out-proj: proj = attn @ out_w^T + out_b : 512x512x512
    gemm_bf16_nt<<<dim3(Dd / 256, NTB / 32, 1), 256, 0, stream>>>(
        attnb, Dd, 0, outw_l, Dd, 0, out_b + (size_t)l * Dd,
        proj, nullptr, Dd, Dd, 2, 4, 0, 0);

    // 7) x = LN(resid + proj)
    add_layernorm<<<NTB, 256, 0, stream>>>(xres, proj,
        ln1_g + (size_t)l * Dd, ln1_b + (size_t)l * Dd, curx, Xb);

    // 8) h = relu(x @ fc1_w^T + fc1_b) -> bf16 : M=512, N=2048, K=512
    gemm_bf16_nt<<<dim3(FFf / 256, NTB / 32, 1), 256, 0, stream>>>(
        Xb, Dd, 0, fc1w_l, Dd, 0, fc1_b + (size_t)l * FFf,
        nullptr, hb, FFf, Dd, 2, 4, /*relu*/1, 0);

    // 9) ffn = h @ fc2_w^T + fc2_b : M=512, N=512, K=2048
    gemm_bf16_nt<<<dim3(Dd / 256, NTB / 32, 1), 256, 0, stream>>>(
        hb, FFf, 0, fc2w_l, FFf, 0, fc2_b + (size_t)l * Dd,
        ffn, nullptr, Dd, FFf, 2, 4, 0, 0);

    // 10) x = LN(x + ffn); final layer writes straight to d_out
    float* dstf = (l == Ll - 1) ? (float*)d_out : curx;
    add_layernorm<<<NTB, 256, 0, stream>>>(curx, ffn,
        ln2_g + (size_t)l * Dd, ln2_b + (size_t)l * Dd, dstf, Xb);
  }
}
